// DeepJ_68315749810523
// MI455X (gfx1250) — matrix operand order repaired and checked by reading, not compiled, not run on hardware
//
#include <hip/hip_runtime.h>
#include <hip/hip_bf16.h>
#include <math.h>

#define BATCH 16384
#define NA    308      // NUM_ACTIONS
#define NS    4        // NUM_STYLES
#define UNITS 512
#define SU    32       // STYLE_UNITS
#define KP0   320      // layer-0 K (308) padded to mult of 32
#define NPAD  320      // output cols padded to mult of 64

#define MB 128         // block M tile
#define NB 64          // block N tile (h columns)
#define KT 32          // K step (WMMA f16 depth)
#define LDS_K (KT + 8) // LDS row stride in halves: 80B, 16B-aligned, conflict-free

typedef __attribute__((ext_vector_type(16))) _Float16 v16h;
typedef __attribute__((ext_vector_type(8)))  float    v8f;
typedef unsigned int u32x4 __attribute__((ext_vector_type(4)));

union Frag { v16h v; _Float16 e[16]; u32x4 q[2]; };
union Acc  { v8f  v; float    e[8];  };

__device__ __forceinline__ float sigmoidf_(float x) { return 1.0f / (1.0f + expf(-x)); }

// Async 16B copy global -> LDS (ASYNCcnt-tracked, no VGPR staging).
// LDS address = low 32 bits of the generic address.
__device__ __forceinline__ void async_copy16(void* lds_ptr, const void* gptr) {
    unsigned lds_addr = (unsigned)(unsigned long long)lds_ptr;
    asm volatile("global_load_async_to_lds_b128 %0, %1, off"
                 :: "v"(lds_addr), "v"(gptr) : "memory");
}
__device__ __forceinline__ void wait_async0() {
    asm volatile("s_wait_asynccnt 0x0" ::: "memory");
}

// ---------------------------------------------------------------------------
// s = style @ Wsl.T + bsl   [B, 32] f32
// ---------------------------------------------------------------------------
__global__ void style_embed_kernel(const float* __restrict__ style,
                                   const float* __restrict__ Wsl,
                                   const float* __restrict__ bsl,
                                   float* __restrict__ s) {
    int i = blockIdx.x * blockDim.x + threadIdx.x;
    if (i >= BATCH * SU) return;
    int b = i >> 5, j = i & 31;
    const float* st = style + b * NS;
    const float* w  = Wsl + j * NS;
    s[i] = bsl[j] + st[0]*w[0] + st[1]*w[1] + st[2]*w[2] + st[3]*w[3];
}

// ---------------------------------------------------------------------------
// Repack LSTM weights: keep gates (i,g,o) only, f16, K padded to Kp.
// ---------------------------------------------------------------------------
__global__ void convert_wih_kernel(const float* __restrict__ Wih, int K, int Kp,
                                   _Float16* __restrict__ W16) {
    long i = (long)blockIdx.x * blockDim.x + threadIdx.x;
    if (i >= 1536L * Kp) return;
    int r = (int)(i / Kp), k = (int)(i % Kp);
    int gate = r >> 9;
    int gbase = (gate == 0) ? 0 : (gate == 1) ? 2 * UNITS : 3 * UNITS;
    float v = (k < K) ? Wih[(size_t)(gbase + (r & 511)) * K + k] : 0.0f;
    W16[i] = (_Float16)v;
}

// Wout [308,512] f32 -> [320,512] f16, zero rows past 308.
__global__ void convert_wout_kernel(const float* __restrict__ W,
                                    _Float16* __restrict__ W16) {
    int i = blockIdx.x * blockDim.x + threadIdx.x;
    if (i >= NPAD * UNITS) return;
    int r = i >> 9, k = i & 511;
    W16[i] = (_Float16)((r < NA) ? W[(size_t)r * UNITS + k] : 0.0f);
}

// ---------------------------------------------------------------------------
// dst[b,j] = f16( Σ residuals + (Ws ? tanh(s[b]·Ws[j]+bs[j]) : 0) ), zero-pad tail.
// ---------------------------------------------------------------------------
__global__ void inject_kernel(const float* __restrict__ p0, int ld0,
                              const float* __restrict__ p1,
                              const float* __restrict__ p2,
                              const float* __restrict__ s,
                              const float* __restrict__ Ws,
                              const float* __restrict__ bs,
                              _Float16* __restrict__ dst,
                              int cols, int stride) {
    __shared__ float sl[SU];
    int b = blockIdx.x;
    if (Ws != nullptr && threadIdx.x < SU) sl[threadIdx.x] = s[b * SU + threadIdx.x];
    __syncthreads();
    for (int j = threadIdx.x; j < stride; j += blockDim.x) {
        float v = 0.0f;
        if (j < cols) {
            if (p0) v += p0[(size_t)b * ld0   + j];
            if (p1) v += p1[(size_t)b * UNITS + j];
            if (p2) v += p2[(size_t)b * UNITS + j];
            if (Ws) {
                const float* w = Ws + (size_t)j * SU;
                float a = bs[j];
#pragma unroll
                for (int k = 0; k < SU; ++k) a += sl[k] * w[k];
                v += tanhf(a);
            }
        }
        dst[(size_t)b * stride + j] = (_Float16)v;
    }
}

// ---------------------------------------------------------------------------
// Fused LSTM gate GEMM (h=c=0): 128x64 block tile, 8 waves of 32x32 x 3 gates.
// Double-buffered LDS filled by global_load_async_to_lds_b128; one barrier/K-step.
//   c = sigmoid(i)*tanh(g);  h = sigmoid(o)*tanh(c)   (f-gate dropped: f*c0 == 0)
// ---------------------------------------------------------------------------
__global__ __launch_bounds__(256)
void lstm_gate_wmma_kernel(const _Float16* __restrict__ A, int lda,
                           const _Float16* __restrict__ W16, int Kp,
                           const float* __restrict__ bih,
                           const float* __restrict__ bhh,
                           float* __restrict__ hOut,
                           float* __restrict__ cOut) {
    __shared__ _Float16 Asl[2][MB][LDS_K];
    __shared__ _Float16 Bsl[2][3][NB][LDS_K];

    const int btile = blockIdx.x, ntile = blockIdx.y;
    const int tid = threadIdx.x, wave = tid >> 5, lane = tid & 31;
    const int mw = wave & 3, nw = wave >> 2;       // 4 M-waves x 2 N-waves
    const int rB = btile * MB;
    const int r0 = mw * 32;
    const int n0 = ntile * NB + nw * 32;

    Acc acc[3][2][2];
#pragma unroll
    for (int g = 0; g < 3; ++g)
#pragma unroll
        for (int mi = 0; mi < 2; ++mi)
#pragma unroll
            for (int ni = 0; ni < 2; ++ni) { v8f z = {}; acc[g][mi][ni].v = z; }

    // Per-thread staging of one K-step: 2 A chunks + 3 B chunks, 16B each, async.
    auto stage = [&](int buf, int k0) {
#pragma unroll
        for (int j = 0; j < 2; ++j) {               // A: 128x32 halves
            int c = j * 256 + tid;
            int row = c >> 2, sfh = (c & 3) * 8;
            async_copy16(&Asl[buf][row][sfh],
                         A + (size_t)(rB + row) * lda + k0 + sfh);
        }
#pragma unroll
        for (int g = 0; g < 3; ++g) {               // B: 3 x 64x32 halves
            int col = tid >> 2, sfh = (tid & 3) * 8;
            int grow = g * UNITS + ntile * NB + col;
            async_copy16(&Bsl[buf][g][col][sfh],
                         W16 + (size_t)grow * Kp + k0 + sfh);
        }
    };

    const int nk = Kp / KT;
    stage(0, 0);
    for (int kk = 0; kk < nk; ++kk) {
        const int buf = kk & 1;
        wait_async0();                 // own stage-kk copies have landed in LDS
        __syncthreads();               // ...now visible to all waves; prior reads done
        if (kk + 1 < nk) stage(buf ^ 1, (kk + 1) * KT);

        Frag af[2];
#pragma unroll
        for (int mi = 0; mi < 2; ++mi) {
            int row = r0 + mi * 16 + (lane & 15);
            int kb = (lane >> 4) * 8;
            af[mi].q[0] = *(const u32x4*)&Asl[buf][row][kb];
            af[mi].q[1] = *(const u32x4*)&Asl[buf][row][16 + kb];
        }
#pragma unroll
        for (int g = 0; g < 3; ++g) {
#pragma unroll
            for (int ni = 0; ni < 2; ++ni) {
                Frag bf;
                int col = nw * 32 + ni * 16 + (lane & 15);
                int kb2 = (lane < 16) ? 0 : 8;
                bf.q[0] = *(const u32x4*)&Bsl[buf][g][col][kb2];
                bf.q[1] = *(const u32x4*)&Bsl[buf][g][col][kb2 + 16];
#pragma unroll
                for (int mi = 0; mi < 2; ++mi)
                    acc[g][mi][ni].v = __builtin_amdgcn_wmma_f32_16x16x32_f16(
                        false, af[mi].v, false, bf.v, (short)0,
                        acc[g][mi][ni].v, false, false);
            }
        }
    }

    // Epilogue: D layout lane -> N = lane&15, M = v + 8*(lane>=16).
    const int colL = lane & 15;
    const int mlo  = (lane >> 4) * 8;
#pragma unroll
    for (int ni = 0; ni < 2; ++ni) {
        int gcol = n0 + ni * 16 + colL;
        float b_i = bih[gcol]           + bhh[gcol];
        float b_g = bih[2*UNITS + gcol] + bhh[2*UNITS + gcol];
        float b_o = bih[3*UNITS + gcol] + bhh[3*UNITS + gcol];
#pragma unroll
        for (int mi = 0; mi < 2; ++mi) {
#pragma unroll
            for (int v = 0; v < 8; ++v) {
                int row = rB + r0 + mi * 16 + mlo + v;
                float gi = sigmoidf_(acc[0][mi][ni].e[v] + b_i);
                float gg = tanhf   (acc[1][mi][ni].e[v] + b_g);
                float go = sigmoidf_(acc[2][mi][ni].e[v] + b_o);
                float c  = gi * gg;
                float h  = go * tanhf(c);
                cOut[(size_t)row * UNITS + gcol] = c;
                hOut[(size_t)row * UNITS + gcol] = h;
            }
        }
    }
}

// ---------------------------------------------------------------------------
// out = xfin @ Wout16.T + bout.  M=B, N=320 (stores guarded to 308), K=512.
// ---------------------------------------------------------------------------
__global__ __launch_bounds__(256)
void out_gemm_wmma_kernel(const _Float16* __restrict__ A,     // [B][512] f16
                          const _Float16* __restrict__ W16,   // [320][512] f16
                          const float* __restrict__ bias,
                          float* __restrict__ out) {
    __shared__ _Float16 Asl[2][MB][LDS_K];
    __shared__ _Float16 Bsl[2][NB][LDS_K];

    const int btile = blockIdx.x, ntile = blockIdx.y;
    const int tid = threadIdx.x, wave = tid >> 5, lane = tid & 31;
    const int mw = wave & 3, nw = wave >> 2;
    const int rB = btile * MB;
    const int r0 = mw * 32;
    const int n0 = ntile * NB + nw * 32;

    Acc acc[2][2];
#pragma unroll
    for (int mi = 0; mi < 2; ++mi)
#pragma unroll
        for (int ni = 0; ni < 2; ++ni) { v8f z = {}; acc[mi][ni].v = z; }

    auto stage = [&](int buf, int k0) {
#pragma unroll
        for (int j = 0; j < 2; ++j) {
            int c = j * 256 + tid;
            int row = c >> 2, sfh = (c & 3) * 8;
            async_copy16(&Asl[buf][row][sfh],
                         A + (size_t)(rB + row) * UNITS + k0 + sfh);
        }
        {
            int col = tid >> 2, sfh = (tid & 3) * 8;
            int grow = ntile * NB + col;
            async_copy16(&Bsl[buf][col][sfh],
                         W16 + (size_t)grow * UNITS + k0 + sfh);
        }
    };

    const int nk = UNITS / KT;
    stage(0, 0);
    for (int kk = 0; kk < nk; ++kk) {
        const int buf = kk & 1;
        wait_async0();
        __syncthreads();
        if (kk + 1 < nk) stage(buf ^ 1, (kk + 1) * KT);

        Frag af[2];
#pragma unroll
        for (int mi = 0; mi < 2; ++mi) {
            int row = r0 + mi * 16 + (lane & 15);
            int kb = (lane >> 4) * 8;
            af[mi].q[0] = *(const u32x4*)&Asl[buf][row][kb];
            af[mi].q[1] = *(const u32x4*)&Asl[buf][row][16 + kb];
        }
#pragma unroll
        for (int ni = 0; ni < 2; ++ni) {
            Frag bf;
            int col = nw * 32 + ni * 16 + (lane & 15);
            int kb2 = (lane < 16) ? 0 : 8;
            bf.q[0] = *(const u32x4*)&Bsl[buf][col][kb2];
            bf.q[1] = *(const u32x4*)&Bsl[buf][col][kb2 + 16];
#pragma unroll
            for (int mi = 0; mi < 2; ++mi)
                acc[mi][ni].v = __builtin_amdgcn_wmma_f32_16x16x32_f16(
                    false, af[mi].v, false, bf.v, (short)0, acc[mi][ni].v, false, false);
        }
    }

    const int colL = lane & 15;
    const int mlo  = (lane >> 4) * 8;
#pragma unroll
    for (int ni = 0; ni < 2; ++ni) {
        int n = n0 + ni * 16 + colL;
        if (n < NA) {
            float b = bias[n];
#pragma unroll
            for (int mi = 0; mi < 2; ++mi)
#pragma unroll
                for (int v = 0; v < 8; ++v) {
                    int row = rB + r0 + mi * 16 + mlo + v;
                    out[(size_t)row * NA + n] = acc[mi][ni].e[v] + b;
                }
        }
    }
}

// ---------------------------------------------------------------------------
extern "C" void kernel_launch(void* const* d_in, const int* in_sizes, int n_in,
                              void* d_out, int out_size, void* d_ws, size_t ws_size,
                              hipStream_t stream) {
    (void)in_sizes; (void)n_in; (void)out_size; (void)ws_size;

    const float* x     = (const float*)d_in[0];
    const float* style = (const float*)d_in[1];
    const float* Wsl   = (const float*)d_in[2];
    const float* bsl   = (const float*)d_in[3];
    const float* Ws0   = (const float*)d_in[4];
    const float* bs0   = (const float*)d_in[5];
    const float* Ws1   = (const float*)d_in[6];
    const float* bs1   = (const float*)d_in[7];
    const float* Ws2   = (const float*)d_in[8];
    const float* bs2   = (const float*)d_in[9];
    const float* Wih0  = (const float*)d_in[10];   // Whh* unused: h==0
    const float* bih0  = (const float*)d_in[12];
    const float* bhh0  = (const float*)d_in[13];
    const float* Wih1  = (const float*)d_in[14];
    const float* bih1  = (const float*)d_in[16];
    const float* bhh1  = (const float*)d_in[17];
    const float* Wih2  = (const float*)d_in[18];
    const float* bih2  = (const float*)d_in[20];
    const float* bhh2  = (const float*)d_in[21];
    const float* Wout  = (const float*)d_in[22];
    const float* bout  = (const float*)d_in[23];

    // d_out: (out, h0, c0, h1, c1, h2, c2) concatenated flat.
    float* out = (float*)d_out;
    float* h0  = out + (size_t)BATCH * NA;
    float* c0  = h0 + (size_t)BATCH * UNITS;
    float* h1  = c0 + (size_t)BATCH * UNITS;
    float* c1  = h1 + (size_t)BATCH * UNITS;
    float* h2  = c1 + (size_t)BATCH * UNITS;
    float* c2  = h2 + (size_t)BATCH * UNITS;

    // Workspace: s f32 [B,32] | xinH f16 [B,512] | W16 f16 [1536, 512]
    char* ws = (char*)d_ws;
    float*     s    = (float*)ws;
    _Float16*  xinH = (_Float16*)(ws + (size_t)BATCH * SU * 4);
    _Float16*  W16  = (_Float16*)(ws + (size_t)BATCH * SU * 4 + (size_t)BATCH * UNITS * 2);

    style_embed_kernel<<<(BATCH * SU + 255) / 256, 256, 0, stream>>>(style, Wsl, bsl, s);

    dim3 ggrid(BATCH / MB, UNITS / NB);   // (128, 8)

    // Layer 0 (K=308, Kp=320)
    convert_wih_kernel<<<(1536 * KP0 + 255) / 256, 256, 0, stream>>>(Wih0, NA, KP0, W16);
    inject_kernel<<<BATCH, 256, 0, stream>>>(x, NA, nullptr, nullptr, s, Ws0, bs0, xinH, NA, KP0);
    lstm_gate_wmma_kernel<<<ggrid, 256, 0, stream>>>(xinH, KP0, W16, KP0, bih0, bhh0, h0, c0);

    // Layer 1 (K=512)
    convert_wih_kernel<<<(1536 * UNITS + 255) / 256, 256, 0, stream>>>(Wih1, UNITS, UNITS, W16);
    inject_kernel<<<BATCH, 256, 0, stream>>>(h0, UNITS, nullptr, nullptr, s, Ws1, bs1, xinH, UNITS, UNITS);
    lstm_gate_wmma_kernel<<<ggrid, 256, 0, stream>>>(xinH, UNITS, W16, UNITS, bih1, bhh1, h1, c1);

    // Layer 2 (input = h0 + h1)
    convert_wih_kernel<<<(1536 * UNITS + 255) / 256, 256, 0, stream>>>(Wih2, UNITS, UNITS, W16);
    inject_kernel<<<BATCH, 256, 0, stream>>>(h0, UNITS, h1, nullptr, s, Ws2, bs2, xinH, UNITS, UNITS);
    lstm_gate_wmma_kernel<<<ggrid, 256, 0, stream>>>(xinH, UNITS, W16, UNITS, bih2, bhh2, h2, c2);

    // Output: xfin = h0 + h1 + h2; out = xfin @ Wout.T + bout
    convert_wout_kernel<<<(NPAD * UNITS + 255) / 256, 256, 0, stream>>>(Wout, W16);
    inject_kernel<<<BATCH, 256, 0, stream>>>(h0, UNITS, h1, h2, nullptr, nullptr, nullptr, xinH, UNITS, UNITS);
    out_gemm_wmma_kernel<<<dim3(BATCH / MB, NPAD / NB), 256, 0, stream>>>(xinH, W16, bout, out);
}
